// FeatureAlign_30803505447002
// MI455X (gfx1250) — compile-verified, matrix operand order checked by reading, not run on hardware
//
#include <hip/hip_runtime.h>

typedef __attribute__((ext_vector_type(16))) __bf16         v16bf;
typedef __attribute__((ext_vector_type(8)))  float          v8f;
typedef __attribute__((ext_vector_type(8)))  unsigned short v8u16;

constexpr int kB      = 4;
constexpr int kC      = 256;   // in & out channels
constexpr int kH      = 64;
constexpr int kW      = 64;
constexpr int kHW     = kH * kW;          // 4096
constexpr int kTaps   = 9;                // 3x3
constexpr int kCK     = kC * kTaps;       // 2304 contraction length
constexpr int kNTILE  = 64;               // one image row per tile
constexpr int kKTILE  = 32;               // bf16 wmma K
constexpr int kGroups = 32;               // group-norm groups
constexpr int kChPerG = kC / kGroups;     // 8
constexpr float kGNInv = 1.0f / (float)(kChPerG * kHW); // 1/32768

__device__ __forceinline__ unsigned short f2bf(float f) {
  unsigned u = __float_as_uint(f);
  unsigned r = u + 0x7FFFu + ((u >> 16) & 1u);   // round-to-nearest-even
  return (unsigned short)(r >> 16);
}

// ---------------------------------------------------------------- stage 1
__global__ void offset_proj_kernel(const float* __restrict__ shape,
                                   const float* __restrict__ w_off,
                                   float* __restrict__ offs) {
  int i = blockIdx.x * blockDim.x + threadIdx.x;
  if (i >= kB * 72 * kHW) return;
  int pix = i & (kHW - 1);
  int bo  = i >> 12;
  int b   = bo / 72;
  int o   = bo - b * 72;
  float acc = 0.f;
#pragma unroll
  for (int ci = 0; ci < 4; ++ci)
    acc += w_off[o * 4 + ci] * shape[((size_t)(b * 4 + ci) << 12) + pix];
  offs[i] = acc;
}

// ---------------------------------------------------------------- stage 2
__global__ void wcvt_kernel(const float* __restrict__ wsrc,
                            unsigned short* __restrict__ wbf) {
  int i = blockIdx.x * blockDim.x + threadIdx.x;
  if (i < kC * kCK) wbf[i] = f2bf(wsrc[i]);
}

__global__ void zero_stats_kernel(float* __restrict__ stats) {
  stats[threadIdx.x] = 0.f;   // 256 floats: (B*32 groups) x {sum, sumsq}
}

// ---------------------------------------------------------------- stage 3
// Fused im2col(bilinear) + bf16 WMMA GEMM + group-stat accumulation.
// One workgroup = one image row (b,h), full M=256; 8 waves x (32M x 64N).
__global__ __launch_bounds__(256)
void deform_gemm_kernel(const float* __restrict__ xin,
                        const float* __restrict__ offs,
                        const unsigned short* __restrict__ wbf,
                        float* __restrict__ out,
                        float* __restrict__ stats) {
  __shared__ __align__(16) unsigned short As[kC * kKTILE];     // 256x32 bf16
  __shared__ __align__(16) unsigned short Bs[kNTILE * kKTILE]; // 64x32  bf16
  __shared__ float4 SampW4[36 * kNTILE];  // masked bilinear weights (corner-validity folded in)
  __shared__ int    SampIdx[36 * kNTILE]; // clamped corner coords packed 4x6 bits
  __shared__ float  ldsSum[kGroups];
  __shared__ float  ldsSq[kGroups];

  const int tid    = threadIdx.x;
  const int lane   = tid & 31;            // wave32
  const int waveId = tid >> 5;            // 8 waves
  const int p0     = blockIdx.x * kNTILE; // global pixel base (one row)
  const int b      = p0 >> 12;
  const int pixrow = p0 & (kHW - 1);
  const int hh     = pixrow >> 6;

  if (tid < kGroups) { ldsSum[tid] = 0.f; ldsSq[tid] = 0.f; }

  // Precompute bilinear sample params for all 36 (dg,tap) x 64 pixels of
  // this row; reused across all 72 K-iterations (64 channels per sample).
  for (int s = tid; s < 36 * kNTILE; s += 256) {
    int dgtap = s >> 6;
    int ww    = s & 63;
    int dg    = dgtap / kTaps;
    int tap   = dgtap - dg * kTaps;
    int ky    = tap / 3;
    int kx    = tap - ky * 3;
    int obase = b * 72 + dg * 18 + tap * 2;
    int pix   = pixrow + ww;
    float dy  = offs[((size_t)obase << 12) + pix];
    float dx  = offs[((size_t)(obase + 1) << 12) + pix];
    float fy  = (float)(hh - 1 + ky) + dy;
    float fx  = (float)(ww - 1 + kx) + dx;
    float fy0 = floorf(fy);
    float fx0 = floorf(fx);
    int iy0 = (int)fy0;
    int ix0 = (int)fx0;
    float wy = fy - fy0;
    float wx = fx - fx0;
    float m0y = (iy0 >= 0 && iy0 < kH) ? 1.f : 0.f;
    float m1y = (iy0 + 1 >= 0 && iy0 + 1 < kH) ? 1.f : 0.f;
    float m0x = (ix0 >= 0 && ix0 < kW) ? 1.f : 0.f;
    float m1x = (ix0 + 1 >= 0 && ix0 + 1 < kW) ? 1.f : 0.f;
    int iy0c = min(max(iy0, 0), kH - 1);
    int iy1c = min(max(iy0 + 1, 0), kH - 1);
    int ix0c = min(max(ix0, 0), kW - 1);
    int ix1c = min(max(ix0 + 1, 0), kW - 1);
    SampIdx[s] = iy0c | (iy1c << 6) | (ix0c << 12) | (ix1c << 18);
    SampW4[s]  = make_float4((1.f - wy) * (1.f - wx) * m0y * m0x,
                             (1.f - wy) * wx         * m0y * m1x,
                             wy * (1.f - wx)         * m1y * m0x,
                             wy * wx                 * m1y * m1x);
  }
  __syncthreads();

  v8f acc[2][4] = {};

  const int mWv   = waveId * 32;          // wave M offset
  const int k0    = (lane & 16) ? 8 : 0;  // ISA 16-bit A/B K-group per lane half
  const int rlane = lane & 15;
  const int mhalf = (lane & 16) ? 8 : 0;

  for (int kt = 0; kt < kCK / kKTILE; ++kt) {
    const int ck0 = kt * kKTILE;

    // --- cooperative A-tile load: 256 rows x 32 bf16 weights (64B/thread)
    {
      const unsigned short* src = wbf + (size_t)tid * kCK + ck0;
      v8u16 t0 = *(const v8u16*)(src);
      v8u16 t1 = *(const v8u16*)(src + 8);
      v8u16 t2 = *(const v8u16*)(src + 16);
      v8u16 t3 = *(const v8u16*)(src + 24);
      *(v8u16*)&As[tid * 32 + 0]  = t0;
      *(v8u16*)&As[tid * 32 + 8]  = t1;
      *(v8u16*)&As[tid * 32 + 16] = t2;
      *(v8u16*)&As[tid * 32 + 24] = t3;
      if (ck0 + kKTILE < kCK)
        __builtin_prefetch(src + kKTILE, 0, 1);  // global_prefetch_b8
    }
    // --- fused im2col B-tile: 64 pixels x 32 ck, branchless bilinear gather
    {
      int kk    = tid & 31;
      int nset  = tid >> 5;
      int ck    = ck0 + kk;
      int c     = ck / kTaps;
      int tap   = ck - c * kTaps;
      int dg    = c >> 6;
      int sbase = (dg * kTaps + tap) * kNTILE;
      const float* xp = xin + ((size_t)(b * kC + c) << 12);
#pragma unroll
      for (int nn = 0; nn < 8; ++nn) {
        int n  = nset * 8 + nn;
        int s  = sbase + n;
        int pk = SampIdx[s];
        float4 wg = SampW4[s];
        int y0s = (pk & 63) << 6;
        int y1s = ((pk >> 6) & 63) << 6;
        int x0i = (pk >> 12) & 63;
        int x1i = (pk >> 18) & 63;
        float v00 = xp[y0s + x0i];
        float v01 = xp[y0s + x1i];
        float v10 = xp[y1s + x0i];
        float v11 = xp[y1s + x1i];
        float val = v00 * wg.x + v01 * wg.y + v10 * wg.z + v11 * wg.w;
        Bs[n * 32 + kk] = f2bf(val);
      }
    }
    __syncthreads();

    // --- fragments per ISA 16-bit layout (two ds_load_b128 each), 8 WMMA
    union Frag { v8u16 u[2]; v16bf bv; };
    Frag a0, a1;
    {
      int ra0 = mWv + rlane, ra1 = mWv + 16 + rlane;
      a0.u[0] = *(const v8u16*)&As[ra0 * 32 + k0];
      a0.u[1] = *(const v8u16*)&As[ra0 * 32 + k0 + 16];
      a1.u[0] = *(const v8u16*)&As[ra1 * 32 + k0];
      a1.u[1] = *(const v8u16*)&As[ra1 * 32 + k0 + 16];
    }
#pragma unroll
    for (int j = 0; j < 4; ++j) {
      Frag bj;
      int rb = j * 16 + rlane;
      bj.u[0] = *(const v8u16*)&Bs[rb * 32 + k0];
      bj.u[1] = *(const v8u16*)&Bs[rb * 32 + k0 + 16];
      acc[0][j] = __builtin_amdgcn_wmma_f32_16x16x32_bf16(false, a0.bv, false, bj.bv, (short)0, acc[0][j], false, false);
      acc[1][j] = __builtin_amdgcn_wmma_f32_16x16x32_bf16(false, a1.bv, false, bj.bv, (short)0, acc[1][j], false, false);
    }
    __syncthreads();
  }

  // --- write conv output + accumulate group-norm stats (pre-reduced over j)
#pragma unroll
  for (int i = 0; i < 2; ++i) {
#pragma unroll
    for (int r = 0; r < 8; ++r) {
      int rowl = mWv + i * 16 + r + mhalf;   // output channel
      float psum = 0.f, psq = 0.f;
#pragma unroll
      for (int j = 0; j < 4; ++j) {
        float val = acc[i][j][r];
        int n = j * 16 + rlane;
        out[((size_t)(b * kC + rowl) << 12) + pixrow + n] = val;
        psum += val;
        psq  += val * val;
      }
      atomicAdd(&ldsSum[rowl >> 3], psum);
      atomicAdd(&ldsSq[rowl >> 3], psq);
    }
  }
  __syncthreads();
  if (tid < kGroups) {
    atomicAdd(&stats[(b * kGroups + tid) * 2 + 0], ldsSum[tid]);
    atomicAdd(&stats[(b * kGroups + tid) * 2 + 1], ldsSq[tid]);
  }
}

// ---------------------------------------------------------------- stage 4
__global__ __launch_bounds__(256)
void gn_relu_kernel(float* __restrict__ out,
                    const float* __restrict__ stats,
                    const float* __restrict__ gamma,
                    const float* __restrict__ beta) {
  int bg = blockIdx.x;
  int b  = bg >> 5;
  int g  = bg & 31;
  float s    = stats[bg * 2 + 0];
  float sq   = stats[bg * 2 + 1];
  float mean = s * kGNInv;
  float var  = sq * kGNInv - mean * mean;
  float rs   = rsqrtf(var + 1e-5f);
  for (int e = threadIdx.x; e < kChPerG * kHW; e += 256) {
    int c_off = e >> 12;
    int pix   = e & (kHW - 1);
    int c     = g * kChPerG + c_off;
    size_t idx = ((size_t)(b * kC + c) << 12) + pix;
    float v = out[idx];
    v = (v - mean) * rs * gamma[c] + beta[c];
    out[idx] = fmaxf(v, 0.f);
  }
}

// ---------------------------------------------------------------- launch
extern "C" void kernel_launch(void* const* d_in, const int* in_sizes, int n_in,
                              void* d_out, int out_size, void* d_ws, size_t ws_size,
                              hipStream_t stream) {
  const float* x     = (const float*)d_in[0];
  const float* shape = (const float*)d_in[1];
  const float* w_off = (const float*)d_in[2];
  const float* w_def = (const float*)d_in[3];
  const float* gamma = (const float*)d_in[4];
  const float* beta  = (const float*)d_in[5];
  float* out = (float*)d_out;

  char* ws = (char*)d_ws;
  float*          offs  = (float*)ws;                                   // 4*72*4096 f32
  unsigned short* wbf   = (unsigned short*)(ws + (size_t)kB * 72 * kHW * 4);
  float*          stats = (float*)(ws + (size_t)kB * 72 * kHW * 4
                                      + (size_t)kC * kCK * 2);          // 256 f32

  offset_proj_kernel<<<(kB * 72 * kHW + 255) / 256, 256, 0, stream>>>(shape, w_off, offs);
  wcvt_kernel<<<(kC * kCK + 255) / 256, 256, 0, stream>>>(w_def, wbf);
  zero_stats_kernel<<<1, 256, 0, stream>>>(stats);

  deform_gemm_kernel<<<(kB * kHW) / kNTILE, 256, 0, stream>>>(x, offs, wbf, out, stats);

  gn_relu_kernel<<<kB * kGroups, 256, 0, stream>>>(out, stats, gamma, beta);
}
